// Filter1Output0_87789131530760
// MI455X (gfx1250) — matrix operand order repaired and checked by reading, not compiled, not run on hardware
//
#include <hip/hip_runtime.h>

// CDNA5 / gfx1250 WMMA implementation of the radial-filter layer.
// out[a,f] = sum_b R[a,b,f] * dot3(rij[a,b,:], L[b,f,:])
// R = relu(rbf @ W1 + b1) @ W2 + b2, computed with v_wmma_f32_16x16x32_f16.

typedef __attribute__((ext_vector_type(16))) _Float16 v16h;
typedef __attribute__((ext_vector_type(8)))  float    v8f;

#define NN   1024
#define RBF  16
#define HID  64
#define FO   32

__global__ __launch_bounds__(256) void filter1_wmma_kernel(
    const float* __restrict__ Lin,   // [1024][32][3]
    const float* __restrict__ rbf,   // [1024][1024][16]
    const float* __restrict__ rij,   // [1024][1024][3]
    const float* __restrict__ W1,    // [16][64]
    const float* __restrict__ b1,    // [64]
    const float* __restrict__ W2,    // [64][32]
    const float* __restrict__ b2,    // [32]
    float* __restrict__ out)         // [1024][32]
{
  // per-wave scratch: H tile (f16, 16x64), S tile (f32, 16x32)
  __shared__ _Float16 hs[8][16 * HID];
  __shared__ float    ss[8][16 * FO];
  __shared__ float    racc0[8][32];
  __shared__ float    racc1[8][32];

  const int a    = blockIdx.x;
  const int tid  = threadIdx.x;
  const int wave = tid >> 5;
  const int lane = tid & 31;
  const int col  = lane & 15;
  const bool hi  = lane >= 16;

  // ---------------- constant B tiles --------------------------------------
  // B order used here, not AMD's table order (32x16 f16): lane<16 -> col=lane, elem e holds K=e (K 0..15);
  //                       lane>=16 -> col=lane-16, elem e holds K=16+e.
  // GEMM1: K padded 16->32, so lanes>=16 are all zero.
  v16h B1t[4];
  #pragma unroll
  for (int t = 0; t < 4; ++t) {
    v16h bb = {};
    if (!hi) {
      #pragma unroll
      for (int i = 0; i < 8; ++i) {
        bb[2 * i]     = (_Float16)W1[(2 * i)     * HID + 16 * t + col];
        bb[2 * i + 1] = (_Float16)W1[(2 * i + 1) * HID + 16 * t + col];
      }
    }
    B1t[t] = bb;
  }

  v16h B2t[2][2];
  #pragma unroll
  for (int kt = 0; kt < 2; ++kt) {
    #pragma unroll
    for (int ft = 0; ft < 2; ++ft) {
      v16h bb = {};
      const int kb = 32 * kt + (hi ? 16 : 0);
      #pragma unroll
      for (int i = 0; i < 8; ++i) {
        bb[2 * i]     = (_Float16)W2[(kb + 2 * i)     * FO + 16 * ft + col];
        bb[2 * i + 1] = (_Float16)W2[(kb + 2 * i + 1) * FO + 16 * ft + col];
      }
      B2t[kt][ft] = bb;
    }
  }

  // C/D layout: lane holds column N = lane%16 -> bias is uniform over the 8 rows
  float bias1[4];
  #pragma unroll
  for (int t = 0; t < 4; ++t) bias1[t] = b1[16 * t + col];
  float bias2[2];
  #pragma unroll
  for (int ft = 0; ft < 2; ++ft) bias2[ft] = b2[16 * ft + col];

  _Float16* hsl = hs[wave];
  float*    ssl = ss[wave];

  float acc0 = 0.f, acc1 = 0.f;

  for (int it = 0; it < 8; ++it) {
    const int b0 = (wave + 8 * it) * 16;

    // ---- S tile: S[b_local][f] = dot3(rij[a,b,:], L[b,f,:]) ----
    {
      const int bl = col;
      const int fb = hi ? 16 : 0;
      const float* rp = rij + ((size_t)a * NN + b0 + bl) * 3;
      const float r0 = rp[0], r1 = rp[1], r2 = rp[2];
      const float* lp = Lin + ((size_t)(b0 + bl) * FO + fb) * 3;
      #pragma unroll
      for (int i = 0; i < 16; ++i)
        ssl[bl * FO + fb + i] = r0 * lp[3 * i] + r1 * lp[3 * i + 1] + r2 * lp[3 * i + 2];
    }

    // ---- A1: rbf tile, 16x16 padded to 16x32 (K 16..31 zero) ----
    // A order used here (contiguous k, matching B): lane<16 row=lane holds K 0..15 in elems 0..15
    //           lane>=16 row=lane-16 holds K 16..31 (pad=0)
    v16h A1 = {};
    {
      const float* src = rbf + ((size_t)a * NN + b0 + col) * RBF;
      #pragma unroll
      for (int i = 0; i < 16; ++i) A1[i] = hi ? (_Float16)0 : (_Float16)src[i];
    }

    // ---- layer 1: 4x WMMA, ReLU, f16 -> LDS (D-layout -> memory re-layout) ----
    #pragma unroll
    for (int t = 0; t < 4; ++t) {
      v8f c;
      #pragma unroll
      for (int e = 0; e < 8; ++e) c[e] = bias1[t];
      c = __builtin_amdgcn_wmma_f32_16x16x32_f16(false, A1, false, B1t[t],
                                                 (short)0, c, false, false);
      const int h0 = 16 * t + col;
      const int mo = hi ? 8 : 0;
      #pragma unroll
      for (int r = 0; r < 8; ++r) {
        float v = c[r];
        v = v > 0.f ? v : 0.f;
        hsl[(mo + r) * HID + h0] = (_Float16)v;
      }
    }

    asm volatile("s_wait_dscnt 0" ::: "memory");

    // ---- A2 tiles (16x64 split into two 16x32 K-chunks) from LDS ----
    v16h A2k0, A2k1;
    {
      const int off = hi ? 16 : 0;
      const _Float16* hp = hsl + col * HID;
      #pragma unroll
      for (int i = 0; i < 8; ++i) {
        A2k0[i]     = hp[off + i];
        A2k0[8 + i] = hp[8 + off + i];
        A2k1[i]     = hp[32 + off + i];
        A2k1[8 + i] = hp[40 + off + i];
      }
    }

    // ---- layer 2 (2 f-tiles, K=64 chained) + contraction with S ----
    #pragma unroll
    for (int ft = 0; ft < 2; ++ft) {
      v8f c;
      #pragma unroll
      for (int e = 0; e < 8; ++e) c[e] = bias2[ft];
      c = __builtin_amdgcn_wmma_f32_16x16x32_f16(false, A2k0, false, B2t[0][ft],
                                                 (short)0, c, false, false);
      c = __builtin_amdgcn_wmma_f32_16x16x32_f16(false, A2k1, false, B2t[1][ft],
                                                 (short)0, c, false, false);
      const int mo = hi ? 8 : 0;
      float part = 0.f;
      #pragma unroll
      for (int r = 0; r < 8; ++r)
        part += c[r] * ssl[(mo + r) * FO + 16 * ft + col];
      if (ft == 0) acc0 += part; else acc1 += part;
    }
  }

  // ---- reduce: halves of the wave, then across the 8 waves ----
  racc0[wave][lane] = acc0;
  racc1[wave][lane] = acc1;
  __syncthreads();

  if (tid < 32) {
    float s = 0.f;
    if (tid < 16) {
      #pragma unroll
      for (int w = 0; w < 8; ++w) s += racc0[w][tid] + racc0[w][tid + 16];
      out[a * FO + tid] = s;
    } else {
      #pragma unroll
      for (int w = 0; w < 8; ++w) s += racc1[w][tid - 16] + racc1[w][tid];
      out[a * FO + tid] = s;
    }
  }
}

extern "C" void kernel_launch(void* const* d_in, const int* in_sizes, int n_in,
                              void* d_out, int out_size, void* d_ws, size_t ws_size,
                              hipStream_t stream) {
  const float* Lin = (const float*)d_in[0];  // layer_inputs [1024,32,3]
  const float* rbf = (const float*)d_in[1];  // rbf_inputs   [1024,1024,16]
  const float* rij = (const float*)d_in[2];  // rij          [1024,1024,3]
  const float* W1  = (const float*)d_in[3];  // [16,64]
  const float* b1  = (const float*)d_in[4];  // [64]
  const float* W2  = (const float*)d_in[5];  // [64,32]
  const float* b2  = (const float*)d_in[6];  // [32]
  float* out = (float*)d_out;                // [1024,32,1]

  filter1_wmma_kernel<<<NN, 256, 0, stream>>>(Lin, rbf, rij, W1, b1, W2, b2, out);
}
